// ModalityCrossAttention_56040733278764
// MI455X (gfx1250) — compile-verified, matrix operand order checked by reading, not run on hardware
//
#include <hip/hip_runtime.h>

typedef __bf16 bf16;
typedef __attribute__((ext_vector_type(16))) __bf16 v16bf;
typedef __attribute__((ext_vector_type(8)))  float  v8f;

#define KKEYS  8192
#define HID    256
#define NHEADS 8
#define HEADD  32
#define BATCH  32

__device__ __forceinline__ float wredMax(float v) {
#pragma unroll
  for (int o = 16; o > 0; o >>= 1) v = fmaxf(v, __shfl_xor(v, o, 32));
  return v;
}
__device__ __forceinline__ float wredSum(float v) {
#pragma unroll
  for (int o = 16; o > 0; o >>= 1) v += __shfl_xor(v, o, 32);
  return v;
}

// ---------------------------------------------------------------------------
// Kernel 1: q-tilde precompute.
//   qh = z_gene @ W_q.T  (per batch row)
//   qt[m][b][n][h] = (0.5 * SCALE * exp(log_temp)) * sum_d qh[n*32+d] * W_k_m[n*32+d, h]
// Folds the entmax z/2 and the score scale into qt so phase 1 of the main
// kernel is a raw dot with kv rows (K/V projection GEMMs are algebraically
// eliminated: scores = (W_k^T qh) . kv,  ctx = W_v (sum_k attn * kv)).
// ---------------------------------------------------------------------------
__global__ void prep_qt_kernel(const float* __restrict__ z_gene,
                               const float* __restrict__ W_q,
                               const float* __restrict__ W_k_cpg,
                               const float* __restrict__ W_k_mirna,
                               const float* __restrict__ log_temp,
                               float* __restrict__ qt) {
  __shared__ float zg[HID];
  __shared__ float qh[HID];
  const int b = blockIdx.x, t = threadIdx.x;
  zg[t] = z_gene[b * HID + t];
  __syncthreads();

  float acc = 0.f;
  const float* wr = W_q + (size_t)t * HID;
#pragma unroll 4
  for (int h = 0; h < HID; ++h) acc += zg[h] * wr[h];
  qh[t] = acc;
  __syncthreads();

  const float cs = 0.5f * rsqrtf((float)HEADD) * __expf(log_temp[0]);
  for (int m = 0; m < 2; ++m) {
    const float* Wk = (m == 0) ? W_k_cpg : W_k_mirna;
    for (int n = 0; n < NHEADS; ++n) {
      float a = 0.f;
#pragma unroll 8
      for (int d = 0; d < HEADD; ++d)
        a += qh[n * HEADD + d] * Wk[(size_t)(n * HEADD + d) * HID + t];
      qt[((size_t)(m * BATCH + b) * NHEADS + n) * HID + t] = a * cs;
    }
  }
}

// ---------------------------------------------------------------------------
// Kernel 2: per (batch, modality) workgroup, 8 waves (wave32), 256 threads.
//   Phase 1: scores[n][k] = qt[n,:] . kv[k,:]  via v_wmma_f32_16x16x32_bf16
//            (M = 16: 8 heads + 8 zero-pad rows; N = 16 keys/tile; 8 K-chunks
//            of 32 over H=256). TWO independent accumulator chains so
//            back-to-back WMMAs have no D->C RAW hazard (avoids v_nop
//            insertion per ISA 7.12.1). Scores -> 256KB dynamic LDS.
//   Phase 2: entmax-1.5 per head row: bisection on tau solving
//            sum(max(z - max - tau, 0)^2) = 1   (z already halved via qt).
//   Phase 3: u[n][h] = (1/sum p) * sum_k p[n][k] * kv[k][h], coalesced loads.
// ---------------------------------------------------------------------------
__global__ void attn_main_kernel(const float* __restrict__ z_cpg,
                                 const float* __restrict__ z_mirna,
                                 const float* __restrict__ qt,
                                 float* __restrict__ u) {
  extern __shared__ float s[];        // NHEADS * KKEYS scores (256 KB)
  __shared__ float invs[NHEADS];

  const int b = blockIdx.x;
  const int m = blockIdx.y;
  const float* kv = (m == 0) ? z_cpg : z_mirna;
  const float* kvb = kv + (size_t)b * KKEYS * HID;

  const int t = threadIdx.x;
  const int w = t >> 5;               // wave id 0..7
  const int lane = t & 31;
  const int hi = lane >> 4;           // lane group (K-half select)
  const int row = lane & 15;          // M row (A) / N col (B)
  const int hioff = hi * 8;

  // ---- Phase 1: WMMA scores ----
  const float* qtb = qt + (size_t)(m * BATCH + b) * NHEADS * HID;
  v16bf afr[8];                       // A fragments, one per 32-wide H chunk
#pragma unroll
  for (int c = 0; c < 8; ++c) {
#pragma unroll
    for (int e = 0; e < 16; ++e) {
      float f = 0.f;
      if (row < NHEADS) {
        const int k = ((e < 8) ? e : e + 8) + hioff;   // 16-bit A K-mapping
        f = qtb[row * HID + c * 32 + k];
      }
      afr[c][e] = (bf16)f;
    }
  }

  // 512 key-tiles of 16; wave w owns keys [w*1024, (w+1)*1024)
  for (int it = 0; it < 64; ++it) {
    const int tbase = (w * 64 + it) * 16;
    const float* kvrow = kvb + (size_t)(tbase + row) * HID;
    // Prefetch next tile's row (global_prefetch_b8) one iteration ahead.
    if (it + 1 < 64)
      __builtin_prefetch(kvrow + 16 * HID, 0, 1);

    v8f acc0 = {0.f, 0.f, 0.f, 0.f, 0.f, 0.f, 0.f, 0.f};
    v8f acc1 = {0.f, 0.f, 0.f, 0.f, 0.f, 0.f, 0.f, 0.f};
#pragma unroll
    for (int c = 0; c < 8; c += 2) {
      v16bf bfr0, bfr1;
      const float* p0 = kvrow + (c + 0) * 32 + hioff;
      const float* p1 = kvrow + (c + 1) * 32 + hioff;
#pragma unroll
      for (int e = 0; e < 16; ++e) {
        const int off = (e < 8) ? e : (e + 8);
        bfr0[e] = (bf16)p0[off];
        bfr1[e] = (bf16)p1[off];
      }
      acc0 = __builtin_amdgcn_wmma_f32_16x16x32_bf16(
          false, afr[c + 0], false, bfr0, (short)0, acc0, false, false);
      acc1 = __builtin_amdgcn_wmma_f32_16x16x32_bf16(
          false, afr[c + 1], false, bfr1, (short)0, acc1, false, false);
    }
    if (hi == 0) {                    // lanes 0..15 hold M rows 0..7 (real heads)
#pragma unroll
      for (int r = 0; r < NHEADS; ++r)
        s[r * KKEYS + tbase + lane] = acc0[r] + acc1[r];
    }
  }
  __syncthreads();

  // ---- Phase 2: entmax-1.5 via bisection; wave w <-> head w ----
  {
    float* zs = s + w * KKEYS;
    float mx = -3.4e38f;
    for (int i = lane; i < KKEYS; i += 32) mx = fmaxf(mx, zs[i]);
    mx = wredMax(mx);

    float lo = -1.0f, hiT = 0.0f;     // tau* in [-1, 0]
    for (int itb = 0; itb < 26; ++itb) {
      const float mid = 0.5f * (lo + hiT);
      float ssum = 0.f;
      for (int i = lane; i < KKEYS; i += 32) {
        float v = fmaxf(zs[i] - mx - mid, 0.f);
        ssum += v * v;
      }
      ssum = wredSum(ssum);
      if (ssum >= 1.0f) lo = mid; else hiT = mid;
    }
    const float tau = 0.5f * (lo + hiT);

    float psum = 0.f;
    for (int i = lane; i < KKEYS; i += 32) {
      const float v = fmaxf(zs[i] - mx - tau, 0.f);
      const float p = v * v;
      zs[i] = p;
      psum += p;
    }
    psum = wredSum(psum);
    if (lane == 0) invs[w] = 1.0f / psum;   // residual renormalization
  }
  __syncthreads();

  // ---- Phase 3: u[n][h] = sum_k attn[n][k] * kv[k][h] ----
  float acc[NHEADS];
#pragma unroll
  for (int n = 0; n < NHEADS; ++n) acc[n] = 0.f;
  const float* kvcol = kvb + t;             // column h = t, coalesced in t
#pragma unroll 4
  for (int k = 0; k < KKEYS; ++k) {
    const float kvv = kvcol[(size_t)k * HID];
#pragma unroll
    for (int n = 0; n < NHEADS; ++n)
      acc[n] += s[n * KKEYS + k] * kvv;     // LDS broadcast
  }
  float* ub = u + (size_t)(m * BATCH + b) * NHEADS * HID;
#pragma unroll
  for (int n = 0; n < NHEADS; ++n)
    ub[n * HID + t] = acc[n] * invs[n];
}

// ---------------------------------------------------------------------------
// Kernel 3: ctx = W_v . u per head, modality-softmax mix, output projection
// + bias + residual, layernorm. One block per batch row.
// ---------------------------------------------------------------------------
__global__ void finish_kernel(const float* __restrict__ u,
                              const float* __restrict__ W_v_cpg,
                              const float* __restrict__ W_v_mirna,
                              const float* __restrict__ W_out,
                              const float* __restrict__ b_out,
                              const float* __restrict__ ln_gamma,
                              const float* __restrict__ ln_beta,
                              const float* __restrict__ modality_logits,
                              const float* __restrict__ z_gene,
                              float* __restrict__ out) {
  __shared__ float uloc[2 * NHEADS * HID];  // 4096 f32
  __shared__ float comb[2 * HID];           // 512 f32
  __shared__ float red[HID];
  const int b = blockIdx.x, t = threadIdx.x;

  for (int i = t; i < 2 * NHEADS * HID; i += HID) {
    const int mm = i / (NHEADS * HID);
    const int r  = i % (NHEADS * HID);
    uloc[i] = u[(size_t)(mm * BATCH + b) * NHEADS * HID + r];
  }
  __syncthreads();

  const float l0 = modality_logits[0], l1 = modality_logits[1];
  const float mx = fmaxf(l0, l1);
  const float e0 = __expf(l0 - mx), e1 = __expf(l1 - mx);
  const float w0 = e0 / (e0 + e1), w1 = e1 / (e0 + e1);

  for (int f = t; f < 2 * HID; f += HID) {
    const int mm = f / HID, r = f % HID, n = r / HEADD;
    const float* Wv = (mm == 0) ? W_v_cpg : W_v_mirna;
    const float* wr = Wv + (size_t)r * HID;
    const float* uv = uloc + (mm * NHEADS + n) * HID;
    float a = 0.f;
#pragma unroll 4
    for (int h = 0; h < HID; ++h) a += uv[h] * wr[h];
    comb[f] = ((mm == 0) ? w0 : w1) * a;
  }
  __syncthreads();

  const float* wrow = W_out + (size_t)t * 2 * HID;
  float y = b_out[t] + z_gene[b * HID + t];
#pragma unroll 4
  for (int j = 0; j < 2 * HID; ++j) y += comb[j] * wrow[j];

  red[t] = y;
  __syncthreads();
  for (int st = HID / 2; st > 0; st >>= 1) {
    if (t < st) red[t] += red[t + st];
    __syncthreads();
  }
  const float mu = red[0] / (float)HID;
  __syncthreads();
  red[t] = (y - mu) * (y - mu);
  __syncthreads();
  for (int st = HID / 2; st > 0; st >>= 1) {
    if (t < st) red[t] += red[t + st];
    __syncthreads();
  }
  const float var = red[0] / (float)HID;
  out[b * HID + t] = (y - mu) * rsqrtf(var + 1e-5f) * ln_gamma[t] + ln_beta[t];
}

// ---------------------------------------------------------------------------
extern "C" void kernel_launch(void* const* d_in, const int* in_sizes, int n_in,
                              void* d_out, int out_size, void* d_ws, size_t ws_size,
                              hipStream_t stream) {
  (void)in_sizes; (void)n_in; (void)out_size; (void)ws_size;
  const float* z_gene = (const float*)d_in[0];
  const float* z_cpg  = (const float*)d_in[1];
  const float* z_mir  = (const float*)d_in[2];
  const float* W_q    = (const float*)d_in[3];
  const float* W_k_c  = (const float*)d_in[4];
  const float* W_v_c  = (const float*)d_in[5];
  const float* W_k_m  = (const float*)d_in[6];
  const float* W_v_m  = (const float*)d_in[7];
  const float* W_outp = (const float*)d_in[8];
  const float* b_outp = (const float*)d_in[9];
  const float* gam    = (const float*)d_in[10];
  const float* bet    = (const float*)d_in[11];
  const float* mlog   = (const float*)d_in[12];
  const float* ltemp  = (const float*)d_in[13];
  float* out = (float*)d_out;

  float* qt = (float*)d_ws;                              // 2*32*8*256 f32
  float* u  = qt + (size_t)2 * BATCH * NHEADS * HID;     // 2*32*8*256 f32

  prep_qt_kernel<<<dim3(BATCH), dim3(HID), 0, stream>>>(
      z_gene, W_q, W_k_c, W_k_m, ltemp, qt);

  attn_main_kernel<<<dim3(BATCH, 2), dim3(256),
                     (size_t)NHEADS * KKEYS * sizeof(float), stream>>>(
      z_cpg, z_mir, qt, u);

  finish_kernel<<<dim3(BATCH), dim3(HID), 0, stream>>>(
      u, W_v_c, W_v_m, W_outp, b_outp, gam, bet, mlog, z_gene, out);
}
